// ParallelBlock_65687229825678
// MI455X (gfx1250) — matrix operand order repaired and checked by reading, not compiled, not run on hardware
//
#include <hip/hip_runtime.h>
#include <hip/hip_bf16.h>

// ---------------------------------------------------------------------------
// Transformer block for MI455X (gfx1250): RMSNorm -> attention + FFN -> resid
// GEMMs + attention einsums via v_wmma_f32_16x16x32_bf16 (wave32 WMMA).
// B-tiles staged to LDS via async-to-LDS copies (ASYNCcnt double buffering).
// Softmax row reductions via DPP (quad_perm / row mirrors), no ds_bpermute.
// ---------------------------------------------------------------------------

typedef __attribute__((ext_vector_type(16))) __bf16 bf16x16;
typedef __attribute__((ext_vector_type(8)))  __bf16 bf16x8;
typedef __attribute__((ext_vector_type(8)))  float  f32x8;

union Frag16 { bf16x16 v; bf16x8 h[2]; };

static __device__ __forceinline__ f32x8 zero8() {
    f32x8 z;
#pragma unroll
    for (int i = 0; i < 8; ++i) z[i] = 0.0f;
    return z;
}

// DPP helper: move with data-parallel-primitive control (16-lane rows)
template <int CTRL>
static __device__ __forceinline__ float fdpp(float x) {
    return __int_as_float(__builtin_amdgcn_update_dpp(
        0, __float_as_int(x), CTRL, 0xF, 0xF, true));
}
// full 16-lane butterfly reductions (rows of 16 == our C-matrix lane groups)
static __device__ __forceinline__ float rmax16(float x) {
    x = fmaxf(x, fdpp<0xB1>(x));    // quad_perm [1,0,3,2]  (xor 1)
    x = fmaxf(x, fdpp<0x4E>(x));    // quad_perm [2,3,0,1]  (xor 2)
    x = fmaxf(x, fdpp<0x141>(x));   // row_half_mirror      (within 8)
    x = fmaxf(x, fdpp<0x140>(x));   // row_mirror           (within 16)
    return x;
}
static __device__ __forceinline__ float rsum16(float x) {
    x += fdpp<0xB1>(x);
    x += fdpp<0x4E>(x);
    x += fdpp<0x141>(x);
    x += fdpp<0x140>(x);
    return x;
}

// async global->LDS copy (16B per lane), tracked by ASYNCcnt.
// Builtin signature (from clang diagnostic): (int4 AS1*, int4 AS3*, i32, i32)
typedef int v4i_b128 __attribute__((vector_size(16)));
static __device__ __forceinline__ void async_copy_b128(__bf16* lds,
                                                       const __bf16* g) {
#if __has_builtin(__builtin_amdgcn_global_load_async_to_lds_b128)
    __builtin_amdgcn_global_load_async_to_lds_b128(
        (__attribute__((address_space(1))) v4i_b128*)g,
        (__attribute__((address_space(3))) v4i_b128*)lds, 0, 0);
#else
    *(float4*)lds = *(const float4*)g;   // fallback: register-staged copy
#endif
}

// Problem constants
#define NTOK 4096   // B*T
#define DM   1024
#define DFF  4096
#define TSEQ 2048
#define NH   16
#define HDIM 64
#define LDB  40     // padded LDS row (bf16 elems): 80B stride, bank-conflict-free

// ---------------------------------------------------------------------------
// RMSNorm: one block per row of 1024, writes bf16 normalized activations
// ---------------------------------------------------------------------------
__global__ __launch_bounds__(256) void rmsnorm_bf16_kernel(
    const float* __restrict__ x, const float* __restrict__ w,
    __bf16* __restrict__ xn)
{
    __shared__ float red[8];
    const int row = blockIdx.x;
    const float4 v = ((const float4*)(x + (size_t)row * DM))[threadIdx.x];
    float ss = v.x * v.x + v.y * v.y + v.z * v.z + v.w * v.w;
#pragma unroll
    for (int m = 1; m < 32; m <<= 1) ss += __shfl_xor(ss, m, 32);
    if ((threadIdx.x & 31) == 0) red[threadIdx.x >> 5] = ss;
    __syncthreads();
    float tot = 0.0f;
#pragma unroll
    for (int i = 0; i < 8; ++i) tot += red[i];
    const float inv = rsqrtf(tot * (1.0f / (float)DM) + 1.1920929e-7f);
    const float4 wv = ((const float4*)w)[threadIdx.x];
    __bf16* out = xn + (size_t)row * DM + threadIdx.x * 4;
    out[0] = (__bf16)(v.x * inv * wv.x);
    out[1] = (__bf16)(v.y * inv * wv.y);
    out[2] = (__bf16)(v.z * inv * wv.z);
    out[3] = (__bf16)(v.w * inv * wv.w);
}

// ---------------------------------------------------------------------------
// Weight transpose + f32 -> bf16:  Wt[m*K + k] = W[k*M + m]
// ---------------------------------------------------------------------------
__global__ __launch_bounds__(256) void transpose_cvt_kernel(
    const float* __restrict__ W, __bf16* __restrict__ Wt, int K, int M)
{
    const int idx = blockIdx.x * 256 + threadIdx.x;
    if (idx >= K * M) return;
    const int m = idx / K;
    const int k = idx - m * K;
    Wt[idx] = (__bf16)W[(size_t)k * M + m];
}

// V [tok, h*64+hd] (bf16) -> Vt [(b*NH+h)*HDIM+hd, t] (bf16)
__global__ __launch_bounds__(256) void vtrans_kernel(
    const __bf16* __restrict__ V, __bf16* __restrict__ Vt)
{
    const int idx = blockIdx.x * 256 + threadIdx.x;   // over NTOK*DM
    const int r = idx >> 11;
    const int t = idx & (TSEQ - 1);
    const int bh = r >> 6;
    const int hd = r & 63;
    const int b = bh >> 4, h = bh & 15;
    Vt[idx] = V[(size_t)(b * TSEQ + t) * DM + h * HDIM + hd];
}

// ---------------------------------------------------------------------------
// WMMA GEMM: C[N x M] = A[N x K] * W, Bt = W^T stored [M x K] bf16.
// 8 waves/block, wave tile 32x64 (8 acc, 8 WMMA / 32-k step).
// B tile (64x32 bf16 = 4KB) staged to LDS by async copy, double buffered.
// EPI: 0 = f32 store, 1 = bf16 store, 2 = SiLU -> bf16 store
// ---------------------------------------------------------------------------
template <int EPI>
__global__ __launch_bounds__(256) void gemm_bf16_wmma(
    const __bf16* __restrict__ A, const __bf16* __restrict__ Bt,
    void* __restrict__ Cout, int Kdim, int Mcols)
{
    __shared__ __bf16 ldsB[2][64 * LDB];
    const int tid  = threadIdx.x;
    const int lane = tid & 31;
    const int w    = tid >> 5;
    const int m16  = lane & 15;
    const int kh   = lane >> 4;
    const int row0 = blockIdx.y * 256 + w * 32;
    const int col0 = blockIdx.x * 64;

    // cooperative copy mapping: 256 threads x 16B = one 64x32 bf16 tile
    const int cCol = tid >> 2;            // 0..63
    const int cK   = (tid & 3) * 8;       // 0,8,16,24
    const __bf16* gB = Bt + (size_t)(col0 + cCol) * Kdim + cK;
    const int lOff = cCol * LDB + cK;

    f32x8 c[8];
#pragma unroll
    for (int j = 0; j < 8; ++j) c[j] = zero8();

    const __bf16* ap0 = A + (size_t)(row0 + m16) * Kdim + kh * 8;
    const __bf16* ap1 = A + (size_t)(row0 + 16 + m16) * Kdim + kh * 8;

    async_copy_b128(&ldsB[0][lOff], gB);

    int buf = 0;
    for (int k = 0; k < Kdim; k += 32, buf ^= 1) {
        if (k + 32 < Kdim) {
            async_copy_b128(&ldsB[buf ^ 1][lOff], gB + k + 32);
            asm volatile("s_wait_asynccnt 1" ::: "memory");
        } else {
            asm volatile("s_wait_asynccnt 0" ::: "memory");
        }
        __syncthreads();   // B tile `buf` visible to all waves

        Frag16 a0, a1;
        a0.h[0] = *(const bf16x8*)(ap0 + k);
        a0.h[1] = *(const bf16x8*)(ap0 + k + 16);
        a1.h[0] = *(const bf16x8*)(ap1 + k);
        a1.h[1] = *(const bf16x8*)(ap1 + k + 16);
#pragma unroll
        for (int j = 0; j < 4; ++j) {
            Frag16 b;
            const __bf16* lp = &ldsB[buf][(j * 16 + m16) * LDB + kh * 8];
            b.h[0] = *(const bf16x8*)lp;
            b.h[1] = *(const bf16x8*)(lp + 16);
            c[j] = __builtin_amdgcn_wmma_f32_16x16x32_bf16(
                false, a0.v, false, b.v, (short)0, c[j], false, false);
            c[4 + j] = __builtin_amdgcn_wmma_f32_16x16x32_bf16(
                false, a1.v, false, b.v, (short)0, c[4 + j], false, false);
        }
        __syncthreads();   // all waves done with `buf` before it is refilled
    }

#pragma unroll
    for (int rb = 0; rb < 2; ++rb) {
#pragma unroll
        for (int j = 0; j < 4; ++j) {
#pragma unroll
            for (int r = 0; r < 8; ++r) {
                const int row = row0 + rb * 16 + kh * 8 + r;
                const int col = col0 + j * 16 + m16;
                const float v = c[rb * 4 + j][r];
                if (EPI == 0) {
                    ((float*)Cout)[(size_t)row * Mcols + col] = v;
                } else if (EPI == 1) {
                    ((__bf16*)Cout)[(size_t)row * Mcols + col] = (__bf16)v;
                } else {
                    const float s = v / (1.0f + __expf(-v));   // SiLU
                    ((__bf16*)Cout)[(size_t)row * Mcols + col] = (__bf16)s;
                }
            }
        }
    }
}

// ---------------------------------------------------------------------------
// Flash attention: one wave per 16-row q-tile (per batch*head).
// Key tiles of 32: S via 4 WMMA, online softmax with DPP row reductions,
// P staged in LDS (bf16, padded rows), O += P@V via 4 WMMA vs transposed V.
// ---------------------------------------------------------------------------
__global__ __launch_bounds__(256) void attn_fwd_kernel(
    const __bf16* __restrict__ Q, const __bf16* __restrict__ Kg,
    const __bf16* __restrict__ Vt, __bf16* __restrict__ ctx)
{
    __shared__ __bf16 pbuf[8][16 * LDB];
    const int lane = threadIdx.x & 31;
    const int w    = threadIdx.x >> 5;
    const int n16  = lane & 15;
    const int kh   = lane >> 4;
    const int bh   = blockIdx.y;              // b*NH + h
    const int b    = bh >> 4, h = bh & 15;
    const int q0   = (blockIdx.x * 8 + w) * 16;
    const int tok0 = b * TSEQ + q0;

    const __bf16* qp = Q + (size_t)(tok0 + n16) * DM + h * HDIM + kh * 16;
    Frag16 aq0, aq1;
    aq0.h[0] = *(const bf16x8*)qp;
    aq0.h[1] = *(const bf16x8*)(qp + 8);
    aq1.h[0] = *(const bf16x8*)(qp + 32);
    aq1.h[1] = *(const bf16x8*)(qp + 40);

    f32x8 o[4];
#pragma unroll
    for (int j = 0; j < 4; ++j) o[j] = zero8();
    float mrow[8], lrow[8], fr[8];
#pragma unroll
    for (int r = 0; r < 8; ++r) { mrow[r] = -INFINITY; lrow[r] = 0.0f; }

    for (int kb = 0; kb <= q0 + 15; kb += 32) {
        f32x8 s0 = zero8(), s1 = zero8();
        {   // S = Q * K^T
            const __bf16* kp =
                Kg + (size_t)(b * TSEQ + kb + n16) * DM + h * HDIM + kh * 16;
            Frag16 bk;
            bk.h[0] = *(const bf16x8*)kp;
            bk.h[1] = *(const bf16x8*)(kp + 8);
            s0 = __builtin_amdgcn_wmma_f32_16x16x32_bf16(
                false, aq0.v, false, bk.v, (short)0, s0, false, false);
            bk.h[0] = *(const bf16x8*)(kp + 32);
            bk.h[1] = *(const bf16x8*)(kp + 40);
            s0 = __builtin_amdgcn_wmma_f32_16x16x32_bf16(
                false, aq1.v, false, bk.v, (short)0, s0, false, false);
            const __bf16* kp2 = kp + (size_t)16 * DM;
            bk.h[0] = *(const bf16x8*)kp2;
            bk.h[1] = *(const bf16x8*)(kp2 + 8);
            s1 = __builtin_amdgcn_wmma_f32_16x16x32_bf16(
                false, aq0.v, false, bk.v, (short)0, s1, false, false);
            bk.h[0] = *(const bf16x8*)(kp2 + 32);
            bk.h[1] = *(const bf16x8*)(kp2 + 40);
            s1 = __builtin_amdgcn_wmma_f32_16x16x32_bf16(
                false, aq1.v, false, bk.v, (short)0, s1, false, false);
        }

        // online softmax update; DPP butterfly over the 16-lane group
#pragma unroll
        for (int r = 0; r < 8; ++r) {
            const int qidx = q0 + kh * 8 + r;
            const float v0 = (kb + n16      <= qidx) ? s0[r] * 0.125f : -INFINITY;
            const float v1 = (kb + 16 + n16 <= qidx) ? s1[r] * 0.125f : -INFINITY;
            const float t    = rmax16(fmaxf(v0, v1));
            const float mnew = fmaxf(mrow[r], t);
            const float f  = __expf(mrow[r] - mnew);
            const float p0 = __expf(v0 - mnew);
            const float p1 = __expf(v1 - mnew);
            lrow[r] = lrow[r] * f + rsum16(p0 + p1);
            mrow[r] = mnew;
            fr[r]   = f;
            const int prow = kh * 8 + r;
            pbuf[w][prow * LDB + n16]      = (__bf16)p0;
            pbuf[w][prow * LDB + 16 + n16] = (__bf16)p1;
        }
#pragma unroll
        for (int j = 0; j < 4; ++j)
#pragma unroll
            for (int r = 0; r < 8; ++r) o[j][r] *= fr[r];

        // lane-crossing read of P within the wave: wait on DS counter
        asm volatile("s_wait_dscnt 0" ::: "memory");

        Frag16 pa;
        const __bf16* pp = &pbuf[w][n16 * LDB + kh * 16];
        pa.h[0] = *(const bf16x8*)pp;
        pa.h[1] = *(const bf16x8*)(pp + 8);

        // O += P @ V
#pragma unroll
        for (int j = 0; j < 4; ++j) {
            const __bf16* vp =
                Vt + (size_t)(bh * 64 + j * 16 + n16) * TSEQ + kb + kh * 16;
            Frag16 bv;
            bv.h[0] = *(const bf16x8*)vp;
            bv.h[1] = *(const bf16x8*)(vp + 8);
            o[j] = __builtin_amdgcn_wmma_f32_16x16x32_bf16(
                false, pa.v, false, bv.v, (short)0, o[j], false, false);
        }
        asm volatile("s_wait_dscnt 0" ::: "memory");
    }

#pragma unroll
    for (int r = 0; r < 8; ++r) {
        const float inv = 1.0f / lrow[r];
        const int tok = tok0 + kh * 8 + r;
#pragma unroll
        for (int j = 0; j < 4; ++j)
            ctx[(size_t)tok * DM + h * HDIM + j * 16 + n16] =
                (__bf16)(o[j][r] * inv);
    }
}

// out = x + attn + ffn
__global__ __launch_bounds__(256) void residual_add_kernel(
    const float* __restrict__ x, const float* __restrict__ a,
    const float* __restrict__ f, float* __restrict__ out, int n)
{
    const int i = blockIdx.x * 256 + threadIdx.x;
    if (i < n) out[i] = x[i] + a[i] + f[i];
}

// ---------------------------------------------------------------------------
// Launch
// ---------------------------------------------------------------------------
extern "C" void kernel_launch(void* const* d_in, const int* in_sizes, int n_in,
                              void* d_out, int out_size, void* d_ws, size_t ws_size,
                              hipStream_t stream)
{
    const float* x  = (const float*)d_in[0];
    const float* nw = (const float*)d_in[1];
    const float* Wq = (const float*)d_in[2];
    const float* Wk = (const float*)d_in[3];
    const float* Wv = (const float*)d_in[4];
    const float* Wo = (const float*)d_in[5];
    const float* W1 = (const float*)d_in[6];
    const float* W2 = (const float*)d_in[7];

    char* ws = (char*)d_ws;
    const size_t SZ_ND_BF  = (size_t)NTOK * DM * 2;        // 8 MB
    const size_t SZ_DD_BF  = (size_t)DM * DM * 2;          // 2 MB
    const size_t SZ_DFF_BF = (size_t)DM * DFF * 2;         // 8 MB
    const size_t SZ_ND_F   = (size_t)NTOK * DM * 4;        // 16 MB

    size_t off = 0;
    __bf16* xn   = (__bf16*)(ws + off); off += SZ_ND_BF;
    __bf16* wq_t = (__bf16*)(ws + off); off += SZ_DD_BF;
    __bf16* wk_t = (__bf16*)(ws + off); off += SZ_DD_BF;
    __bf16* wv_t = (__bf16*)(ws + off); off += SZ_DD_BF;
    __bf16* wo_t = (__bf16*)(ws + off); off += SZ_DD_BF;
    __bf16* w1_t = (__bf16*)(ws + off); off += SZ_DFF_BF;  // [DFF x DM]
    __bf16* w2_t = (__bf16*)(ws + off); off += SZ_DFF_BF;  // [DM x DFF]
    __bf16* qb   = (__bf16*)(ws + off); off += SZ_ND_BF;
    __bf16* kb   = (__bf16*)(ws + off); off += SZ_ND_BF;
    __bf16* vb   = (__bf16*)(ws + off); off += SZ_ND_BF;
    __bf16* vtb  = (__bf16*)(ws + off); off += SZ_ND_BF;
    float*  attn = (float*)(ws + off);  off += SZ_ND_F;
    float*  ffn  = (float*)(ws + off);  off += SZ_ND_F;
    __bf16* ctx  = vb;                 // V bf16 dead after vtrans
    __bf16* h1   = qb;                 // Q..Vt (32MB) reused for SiLU hidden

    rmsnorm_bf16_kernel<<<NTOK, 256, 0, stream>>>(x, nw, xn);

    transpose_cvt_kernel<<<(DM * DM) / 256, 256, 0, stream>>>(Wq, wq_t, DM, DM);
    transpose_cvt_kernel<<<(DM * DM) / 256, 256, 0, stream>>>(Wk, wk_t, DM, DM);
    transpose_cvt_kernel<<<(DM * DM) / 256, 256, 0, stream>>>(Wv, wv_t, DM, DM);
    transpose_cvt_kernel<<<(DM * DM) / 256, 256, 0, stream>>>(Wo, wo_t, DM, DM);
    transpose_cvt_kernel<<<(DM * DFF) / 256, 256, 0, stream>>>(W1, w1_t, DM, DFF);
    transpose_cvt_kernel<<<(DM * DFF) / 256, 256, 0, stream>>>(W2, w2_t, DFF, DM);

    dim3 gDD(DM / 64, NTOK / 256);        // (16, 16)
    gemm_bf16_wmma<1><<<gDD, 256, 0, stream>>>(xn, wq_t, qb, DM, DM);
    gemm_bf16_wmma<1><<<gDD, 256, 0, stream>>>(xn, wk_t, kb, DM, DM);
    gemm_bf16_wmma<1><<<gDD, 256, 0, stream>>>(xn, wv_t, vb, DM, DM);
    vtrans_kernel<<<(NTOK * DM) / 256, 256, 0, stream>>>(vb, vtb);

    attn_fwd_kernel<<<dim3(16, 32), 256, 0, stream>>>(qb, kb, vtb, ctx);

    gemm_bf16_wmma<0><<<gDD, 256, 0, stream>>>(ctx, wo_t, attn, DM, DM);

    dim3 gFF(DFF / 64, NTOK / 256);       // (64, 16)
    gemm_bf16_wmma<2><<<gFF, 256, 0, stream>>>(xn, w1_t, h1, DM, DFF);
    gemm_bf16_wmma<0><<<gDD, 256, 0, stream>>>(h1, w2_t, ffn, DFF, DM);

    residual_add_kernel<<<(NTOK * DM) / 256, 256, 0, stream>>>(
        x, attn, ffn, (float*)d_out, NTOK * DM);
}
